// GeometricCore_33629593927926
// MI455X (gfx1250) — matrix operand order repaired and checked by reading, not compiled, not run on hardware
//
#include <hip/hip_runtime.h>

typedef __attribute__((ext_vector_type(16))) _Float16 v16h;
typedef __attribute__((ext_vector_type(8)))  float    v8f;
typedef __attribute__((ext_vector_type(4)))  float    f32x4;

#define NPROTO          32
#define DIM             64
#define INV_T           0.2f          // 1 / SOFTMAX_TEMP(5.0)
#define WAVES_PER_BLOCK 4
#define ROWS_PER_WAVE   16
#define ROWS_PER_BLOCK  (WAVES_PER_BLOCK * ROWS_PER_WAVE)   // 64; B=1e6 -> 15625 blocks exactly

// ---------------------------------------------------------------- init: zero scratch histogram
__global__ void basins_init_hist(unsigned* __restrict__ hist) {
    if (threadIdx.x < NPROTO) hist[threadIdx.x] = 0u;
}

// ---------------------------------------------------------------- main kernel
__global__ __launch_bounds__(128) void basins_main(
        const float* __restrict__ z,        // [n, 64]
        const float* __restrict__ protos,   // [32, 64]
        float*       __restrict__ act,      // [n, 32]
        float*       __restrict__ assign,   // [n]
        unsigned*    __restrict__ hist,     // [32] scratch histogram
        int n)
{
    __shared__ __align__(16) float lds_z [WAVES_PER_BLOCK][ROWS_PER_WAVE * DIM];    // 16 KB
    __shared__ __align__(16) float lds_s [WAVES_PER_BLOCK][ROWS_PER_WAVE * NPROTO]; //  8 KB
    __shared__ float    lds_max [WAVES_PER_BLOCK][ROWS_PER_WAVE];
    __shared__ float    lds_rinv[WAVES_PER_BLOCK][ROWS_PER_WAVE];
    __shared__ int      lds_idx [WAVES_PER_BLOCK][ROWS_PER_WAVE][3];
    __shared__ unsigned lds_hist[NPROTO];

    const int tid  = threadIdx.x;
    const int wave = tid >> 5;
    const int lane = tid & 31;
    const int lr   = lane & 15;   // M row (A/C) or N column (B/C) within 16
    const int hi   = lane >> 4;   // lane-half selector

    if (tid < NPROTO) lds_hist[tid] = 0u;

    const int rowBase = (blockIdx.x * WAVES_PER_BLOCK + wave) * ROWS_PER_WAVE;
    // n is a multiple of ROWS_PER_BLOCK for this workload (B = 1,000,000).

    // ---- B operands: prototypes^T as WMMA B operand (K x N), contiguous K order, f32 -> f16 ----
    // lane (hi,lr): column N = lr (+16*nh), holds K = hi*16 .. hi*16+15 (+32*ks)
    v16h bmat[2][2];
    #pragma unroll
    for (int nh = 0; nh < 2; ++nh) {
        #pragma unroll
        for (int ks = 0; ks < 2; ++ks) {
            const float* src = protos + (nh * 16 + lr) * DIM + ks * 32 + hi * 16;
            f32x4 p0 = *(const f32x4*)(src);
            f32x4 p1 = *(const f32x4*)(src + 4);
            f32x4 p2 = *(const f32x4*)(src + 8);
            f32x4 p3 = *(const f32x4*)(src + 12);
            v16h b;
            #pragma unroll
            for (int i = 0; i < 4; ++i) {
                b[i]      = (_Float16)p0[i];
                b[4 + i]  = (_Float16)p1[i];
                b[8 + i]  = (_Float16)p2[i];
                b[12 + i] = (_Float16)p3[i];
            }
            bmat[nh][ks] = b;
        }
    }

    // ---- coalesced z tile (16 rows x 64 f32 = 256 f32x4) into LDS ----
    {
        const f32x4* zg = (const f32x4*)(z + (size_t)rowBase * DIM);
        f32x4*       zl = (f32x4*)lds_z[wave];
        #pragma unroll
        for (int i = 0; i < 8; ++i)
            zl[i * 32 + lane] = zg[i * 32 + lane];
    }
    __syncthreads();

    // ---- A operands from LDS (contiguous K order, same as B) + 4x WMMA: C = z * protos^T ----
    // lane half 0: K 0-15; lane half 1: K 16-31 (+32*ks)
    v8f c0 = {};   // protos 0..15
    v8f c1 = {};   // protos 16..31
    #pragma unroll
    for (int ks = 0; ks < 2; ++ks) {
        const float* zr = lds_z[wave] + lr * DIM + ks * 32 + hi * 16;
        const f32x4* q  = (const f32x4*)zr;      // 64B-aligned
        f32x4 a0 = q[0], a1 = q[1];              // K +0..7
        f32x4 a2 = q[2], a3 = q[3];              // K +8..15
        v16h a;
        #pragma unroll
        for (int i = 0; i < 4; ++i) {
            a[i]      = (_Float16)a0[i];
            a[4 + i]  = (_Float16)a1[i];
            a[8 + i]  = (_Float16)a2[i];
            a[12 + i] = (_Float16)a3[i];
        }
        c0 = __builtin_amdgcn_wmma_f32_16x16x32_f16(false, a, false, bmat[0][ks],
                                                    (short)0, c0, false, false);
        c1 = __builtin_amdgcn_wmma_f32_16x16x32_f16(false, a, false, bmat[1][ks],
                                                    (short)0, c1, false, false);
    }

    // ---- scatter sims to LDS: C layout VGPR r -> M = hi*8 + r, N = lr ----
    {
        const int mb = hi * 8;
        #pragma unroll
        for (int r = 0; r < 8; ++r) {
            lds_s[wave][(mb + r) * NPROTO + lr]      = c0[r];
            lds_s[wave][(mb + r) * NPROTO + 16 + lr] = c1[r];
        }
    }
    __syncthreads();

    // ---- per-row top-3 + softmax params (lanes 0..15, one row each) ----
    if (lane < ROWS_PER_WAVE) {
        const float* srow = lds_s[wave] + lane * NPROTO;
        float v0 = -3.0e38f, v1 = -3.0e38f, v2 = -3.0e38f;
        int   i0 = 0, i1 = 0, i2 = 0;
        #pragma unroll
        for (int j = 0; j < NPROTO; ++j) {
            float v = srow[j];
            if (v > v0)      { v2 = v1; i2 = i1; v1 = v0; i1 = i0; v0 = v; i0 = j; }
            else if (v > v1) { v2 = v1; i2 = i1; v1 = v;  i1 = j; }
            else if (v > v2) { v2 = v;  i2 = j; }
        }
        // masked softmax: non-top entries are exp(NEG_BIG/T) == 0 in f32
        float denom = 1.0f + __expf((v1 - v0) * INV_T) + __expf((v2 - v0) * INV_T);
        lds_max [wave][lane]    = v0;
        lds_rinv[wave][lane]    = 1.0f / denom;
        lds_idx [wave][lane][0] = i0;
        lds_idx [wave][lane][1] = i1;
        lds_idx [wave][lane][2] = i2;
        assign[rowBase + lane] = (float)i0;        // top_idx[:,0]
        atomicAdd(&lds_hist[i0], 1u);
    }
    __syncthreads();

    // ---- activation tile, fully coalesced (16x32 f32 per wave) ----
    {
        float* arow = act + (size_t)rowBase * NPROTO;
        #pragma unroll
        for (int i = 0; i < 16; ++i) {
            int   t = i * 32 + lane;
            int   r = t >> 5, cidx = t & 31;
            float s = lds_s[wave][t];
            int   j0 = lds_idx[wave][r][0];
            int   j1 = lds_idx[wave][r][1];
            int   j2 = lds_idx[wave][r][2];
            float o = 0.0f;
            if (cidx == j0 || cidx == j1 || cidx == j2)
                o = __expf((s - lds_max[wave][r]) * INV_T) * lds_rinv[wave][r];
            arow[t] = o;
        }
    }

    __syncthreads();
    if (tid < NPROTO) atomicAdd(&hist[tid], lds_hist[tid]);
}

// ---------------------------------------------------------------- finalize utilization
__global__ void basins_finalize(const float* __restrict__ util_in,
                                const unsigned* __restrict__ hist,
                                float* __restrict__ util_out) {
    int i = threadIdx.x;
    if (i < NPROTO) util_out[i] = util_in[i] + (float)hist[i];
}

// ---------------------------------------------------------------- launch
extern "C" void kernel_launch(void* const* d_in, const int* in_sizes, int n_in,
                              void* d_out, int out_size, void* d_ws, size_t ws_size,
                              hipStream_t stream) {
    const float* z      = (const float*)d_in[0];   // [B,64]
    const float* protos = (const float*)d_in[1];   // [32,64]
    const float* util   = (const float*)d_in[2];   // [32]
    const int n = in_sizes[0] / DIM;               // B = 1,000,000

    float* act      = (float*)d_out;               // [B,32]
    float* assign   = act + (size_t)n * NPROTO;    // [B]
    float* util_out = assign + n;                  // [32]
    unsigned* hist  = (unsigned*)d_ws;             // [32] scratch

    basins_init_hist<<<1, 32, 0, stream>>>(hist);
    basins_main<<<n / ROWS_PER_BLOCK, 128, 0, stream>>>(z, protos, act, assign, hist, n);
    basins_finalize<<<1, 32, 0, stream>>>(util, hist, util_out);
}
